// Idefics3SelectorConnector_22754736734719
// MI455X (gfx1250) — matrix operand order repaired and checked by reading, not compiled, not run on hardware
//
#include <hip/hip_runtime.h>
#include <hip/hip_bf16.h>
#include <cmath>

// ---------- CDNA5 WMMA plumbing (wave32, 16x16x32 bf16 -> f32) ----------
typedef __attribute__((ext_vector_type(16))) __bf16 bf16x16;
typedef __attribute__((ext_vector_type(8)))  __bf16 bf16x8;
typedef __attribute__((ext_vector_type(8)))  float  f32x8;

union BF16Frag { bf16x16 v; bf16x8 h[2]; };

__device__ __forceinline__ f32x8 wmma_bf16(bf16x16 a, bf16x16 b, f32x8 c) {
  return __builtin_amdgcn_wmma_f32_16x16x32_bf16(false, a, false, b, (short)0, c, false, false);
}

// A-matrix fragment (16x32, MxK): lane holds row M=lane&15.
// halves 0-7 : K = half*8 + h ; halves 8-15 : K = 16 + half*8 + (h-8)
__device__ __forceinline__ bf16x16 load_fragA(const __bf16* rowBase, int halfId, int fbase) {
  BF16Frag f;
  f.h[0] = *(const bf16x8*)(rowBase + fbase + halfId * 8);
  f.h[1] = *(const bf16x8*)(rowBase + fbase + halfId * 8 + 16);
  return f.v;
}
// B-matrix fragment (32x16, KxN): lane holds column N=lane&15.
// halves 0-7 : K = half*8 + h ; halves 8-15 : K = 16 + half*8 + (h-8)
__device__ __forceinline__ bf16x16 load_fragB(const __bf16* colBase, int halfId, int fbase) {
  BF16Frag f;
  f.h[0] = *(const bf16x8*)(colBase + fbase + halfId * 8);
  f.h[1] = *(const bf16x8*)(colBase + fbase + halfId * 8 + 16);
  return f.v;
}

// Async bf16 tile copy: global (16B chunk) -> LDS, tracked by ASYNCcnt.
__device__ __forceinline__ void async_copy_b128(__bf16* ldsDst, const __bf16* gSrc) {
  unsigned l = (unsigned)(unsigned long long)ldsDst; // low 32 bits = LDS offset
  asm volatile("global_load_async_to_lds_b128 %0, %1, off" :: "v"(l), "v"(gSrc) : "memory");
}
__device__ __forceinline__ void async_wait0() {
  asm volatile("s_wait_asynccnt 0" ::: "memory");
}

// ---------- f32->bf16 conversion ----------
__global__ void k_cvt(const float* __restrict__ src, __bf16* __restrict__ dst, long n)
{
  long i = (long)blockIdx.x * blockDim.x + threadIdx.x;
  if (i < n) dst[i] = (__bf16)src[i];
}

// ---------- big GEMM (bf16 operands): C[M,N] = A[M,K] @ W[N,K]^T + bias ----------
// M%128==0, N%64==0, K%32==0. Block 256 thr (8 waves); tile 128x64; wave does 2x2 WMMA.
// Tiles are staged through LDS with global_load_async_to_lds_b128 (ASYNCcnt).
__global__ __launch_bounds__(256)
void k_gemm_bf16(const __bf16* __restrict__ A, const __bf16* __restrict__ W,
                 const float* __restrict__ bias, float* __restrict__ C, int N, int K)
{
  __shared__ __align__(16) __bf16 As[128 * 40];
  __shared__ __align__(16) __bf16 Ws[64 * 40];
  const int t = threadIdx.x;
  const int lane = t & 31;
  const int col = lane & 15;
  const int halfId = lane >> 4;
  const int w = t >> 5;
  const long m0 = (long)blockIdx.y * 128;
  const long n0 = (long)blockIdx.x * 64;
  const int msub = (w & 3) * 32;   // wave covers 32 rows (2 A frags)
  const int nsub = (w >> 2) * 32;  // wave covers 32 cols (2 B frags)

  f32x8 acc00 = {}, acc01 = {}, acc10 = {}, acc11 = {};

  for (int k0 = 0; k0 < K; k0 += 32) {
    // 768 16-byte chunks per stage: 512 for A (128x32), 256 for W (64x32)
#pragma unroll
    for (int j = 0; j < 3; ++j) {
      int idx = t + j * 256;
      if (idx < 512) {
        int row = idx >> 2, c = idx & 3;
        async_copy_b128(As + row * 40 + c * 8, A + (m0 + row) * (long)K + k0 + c * 8);
      } else {
        int idx2 = idx - 512;
        int row = idx2 >> 2, c = idx2 & 3;
        async_copy_b128(Ws + row * 40 + c * 8, W + (n0 + row) * (long)K + k0 + c * 8);
      }
    }
    async_wait0();
    __syncthreads();
    bf16x16 fa0 = load_fragA(As + (msub + col) * 40, halfId, 0);
    bf16x16 fa1 = load_fragA(As + (msub + 16 + col) * 40, halfId, 0);
    bf16x16 fb0 = load_fragB(Ws + (nsub + col) * 40, halfId, 0);
    bf16x16 fb1 = load_fragB(Ws + (nsub + 16 + col) * 40, halfId, 0);
    acc00 = wmma_bf16(fa0, fb0, acc00);
    acc01 = wmma_bf16(fa0, fb1, acc01);
    acc10 = wmma_bf16(fa1, fb0, acc10);
    acc11 = wmma_bf16(fa1, fb1, acc11);
    __syncthreads();
  }

  const long nc0 = n0 + nsub + col;
  const long nc1 = nc0 + 16;
  const float b0 = bias ? bias[nc0] : 0.f;
  const float b1 = bias ? bias[nc1] : 0.f;
#pragma unroll
  for (int r = 0; r < 8; ++r) {
    long ma = m0 + msub + halfId * 8 + r;
    long mb = ma + 16;
    C[ma * N + nc0] = acc00[r] + b0;
    C[ma * N + nc1] = acc01[r] + b1;
    C[mb * N + nc0] = acc10[r] + b0;
    C[mb * N + nc1] = acc11[r] + b1;
  }
}

// ---------- small GEMM (f32 in, fused bias/resid/gelu): tile 64x64 ----------
__global__ __launch_bounds__(256)
void k_gemm(const float* __restrict__ A, const float* __restrict__ W,
            const float* __restrict__ bias, const float* __restrict__ resid,
            float* __restrict__ C, int M, int N, int K, int do_gelu)
{
  (void)M;
  __shared__ __align__(16) __bf16 As[64 * 40];
  __shared__ __align__(16) __bf16 Ws[64 * 40];
  const int t = threadIdx.x;
  const int lane = t & 31;
  const int col = lane & 15;
  const int halfId = lane >> 4;
  const int w = t >> 5;
  const long m0 = (long)blockIdx.y * 64;
  const long n0 = (long)blockIdx.x * 64;
  const int msub = (w & 3) * 16;
  const int nsub0 = ((w >> 2) * 2) * 16;
  const int nsub1 = nsub0 + 16;

  f32x8 acc0 = {}; f32x8 acc1 = {};

  for (int k0 = 0; k0 < K; k0 += 32) {
    if (k0 + 32 < K) {
      __builtin_prefetch(A + (m0 + (t >> 3)) * K + k0 + 32, 0, 1);
      __builtin_prefetch(W + (n0 + (t >> 3)) * K + k0 + 32, 0, 1);
    }
#pragma unroll
    for (int i = 0; i < 2; ++i) {
      int idx = t + i * 256;
      int row = idx >> 3;
      int c4 = (idx & 7) * 4;
      float4 a4 = *(const float4*)(A + (m0 + row) * K + k0 + c4);
      float4 w4 = *(const float4*)(W + (n0 + row) * K + k0 + c4);
      __bf16* ap = As + row * 40 + c4;
      ap[0] = (__bf16)a4.x; ap[1] = (__bf16)a4.y; ap[2] = (__bf16)a4.z; ap[3] = (__bf16)a4.w;
      __bf16* wp = Ws + row * 40 + c4;
      wp[0] = (__bf16)w4.x; wp[1] = (__bf16)w4.y; wp[2] = (__bf16)w4.z; wp[3] = (__bf16)w4.w;
    }
    __syncthreads();
    bf16x16 fa  = load_fragA(As + (msub + col) * 40, halfId, 0);
    bf16x16 fb0 = load_fragB(Ws + (nsub0 + col) * 40, halfId, 0);
    bf16x16 fb1 = load_fragB(Ws + (nsub1 + col) * 40, halfId, 0);
    acc0 = wmma_bf16(fa, fb0, acc0);
    acc1 = wmma_bf16(fa, fb1, acc1);
    __syncthreads();
  }

  const long mrow = m0 + msub + halfId * 8;
  const long nc0 = n0 + nsub0 + col;
  const long nc1 = n0 + nsub1 + col;
  const float b0 = bias ? bias[nc0] : 0.f;
  const float b1 = bias ? bias[nc1] : 0.f;
#pragma unroll
  for (int r = 0; r < 8; ++r) {
    long m = mrow + r;
    float v0 = acc0[r] + b0;
    float v1 = acc1[r] + b1;
    if (resid) { v0 += resid[m * N + nc0]; v1 += resid[m * N + nc1]; }
    if (do_gelu) {
      v0 = 0.5f * v0 * (1.f + erff(v0 * 0.70710678f));
      v1 = 0.5f * v1 * (1.f + erff(v1 * 0.70710678f));
    }
    C[m * N + nc0] = v0;
    C[m * N + nc1] = v1;
  }
}

// ---------- pack heads to bf16 [R=B*H, Nq, 96] (dh=72 zero-padded to 96) ----------
__global__ void k_pack(const float* __restrict__ src, __bf16* __restrict__ dst,
                       int H, int Nq, long srcRowStride, long srcBatchStride,
                       int colOff, float scale, long total)
{
  long i = (long)blockIdx.x * blockDim.x + threadIdx.x;
  if (i >= total) return;
  int d = (int)(i % 96);
  long rest = i / 96;
  int q = (int)(rest % Nq);
  long r = rest / Nq;
  int h = (int)(r % H);
  long b = r / H;
  float v = 0.f;
  if (d < 72) v = src[b * srcBatchStride + (long)q * srcRowStride + colOff + h * 72 + d] * scale;
  dst[i] = (__bf16)v;
}

// ---------- pack V transposed: [R, 96, Nk] so B-fragments are contiguous ----------
__global__ void k_pack_vT(const float* __restrict__ src, __bf16* __restrict__ dst,
                          int H, int Nk, long srcRowStride, long srcBatchStride,
                          int colOff, long total)
{
  long i = (long)blockIdx.x * blockDim.x + threadIdx.x;
  if (i >= total) return;
  int kk = (int)(i % Nk);
  long rest = i / Nk;
  int d = (int)(rest % 96);
  long r = rest / 96;
  int h = (int)(r % H);
  long b = r / H;
  float v = 0.f;
  if (d < 72) v = src[b * srcBatchStride + (long)kk * srcRowStride + colOff + h * 72 + d];
  dst[i] = (__bf16)v;
}

// ---------- flash max/sumexp stats per query row ----------
__global__ __launch_bounds__(32)
void k_attn_stats(const __bf16* __restrict__ Qp, const __bf16* __restrict__ Kp,
                  float* __restrict__ stats, int Nq, int Nk)
{
  const int lane = threadIdx.x;
  const int col = lane & 15;
  const int halfId = lane >> 4;
  const long r = blockIdx.y;
  const int qt = blockIdx.x;

  bf16x16 qa[3];
  const __bf16* qrow = Qp + ((r * Nq) + qt * 16 + col) * 96;
#pragma unroll
  for (int f = 0; f < 3; ++f) qa[f] = load_fragA(qrow, halfId, f * 32);

  float mrun[8], lrun[8];
#pragma unroll
  for (int j = 0; j < 8; ++j) { mrun[j] = -3.0e38f; lrun[j] = 0.f; }

  for (int kt = 0; kt < Nk; kt += 16) {
    const __bf16* krow = Kp + ((r * Nk) + kt + col) * 96;
    f32x8 c = {};
#pragma unroll
    for (int f = 0; f < 3; ++f) c = wmma_bf16(qa[f], load_fragB(krow, halfId, f * 32), c);
#pragma unroll
    for (int j = 0; j < 8; ++j) {
      float tm = c[j];
      tm = fmaxf(tm, __shfl_xor(tm, 1, 32));
      tm = fmaxf(tm, __shfl_xor(tm, 2, 32));
      tm = fmaxf(tm, __shfl_xor(tm, 4, 32));
      tm = fmaxf(tm, __shfl_xor(tm, 8, 32));
      float mnew = fmaxf(mrun[j], tm);
      float e = __expf(c[j] - mnew);
      e += __shfl_xor(e, 1, 32);
      e += __shfl_xor(e, 2, 32);
      e += __shfl_xor(e, 4, 32);
      e += __shfl_xor(e, 8, 32);
      lrun[j] = lrun[j] * __expf(mrun[j] - mnew) + e;
      mrun[j] = mnew;
    }
  }
  if (col == 0) {
#pragma unroll
    for (int j = 0; j < 8; ++j) {
      long q = (long)qt * 16 + halfId * 8 + j;
      stats[(r * Nq + q) * 2 + 0] = mrun[j];
      stats[(r * Nq + q) * 2 + 1] = lrun[j];
    }
  }
}

// ---------- selector: accumulate s2[b,k] = sum_{h,q} softmax probs ----------
__global__ __launch_bounds__(32)
void k_sel_s2(const __bf16* __restrict__ Qp, const __bf16* __restrict__ Kp,
              const float* __restrict__ stats, float* __restrict__ s2g,
              int Nq, int Nk, int H)
{
  const int lane = threadIdx.x;
  const int col = lane & 15;
  const int halfId = lane >> 4;
  const long r = blockIdx.y;
  const int qt = blockIdx.x;
  const long b = r / H;

  bf16x16 qa[3];
  const __bf16* qrow = Qp + ((r * Nq) + qt * 16 + col) * 96;
#pragma unroll
  for (int f = 0; f < 3; ++f) qa[f] = load_fragA(qrow, halfId, f * 32);

  float mx[8], il[8];
#pragma unroll
  for (int j = 0; j < 8; ++j) {
    long q = (long)qt * 16 + halfId * 8 + j;
    mx[j] = stats[(r * Nq + q) * 2 + 0];
    il[j] = 1.f / stats[(r * Nq + q) * 2 + 1];
  }
  for (int kt = 0; kt < Nk; kt += 16) {
    const __bf16* krow = Kp + ((r * Nk) + kt + col) * 96;
    f32x8 c = {};
#pragma unroll
    for (int f = 0; f < 3; ++f) c = wmma_bf16(qa[f], load_fragB(krow, halfId, f * 32), c);
    float cs = 0.f;
#pragma unroll
    for (int j = 0; j < 8; ++j) cs += __expf(c[j] - mx[j]) * il[j];
    cs += __shfl_xor(cs, 16, 32);
    if (lane < 16) atomicAdd(&s2g[b * Nk + kt + col], cs);
  }
}

// ---------- attention output: O[R,Nq,96] = softmax(QK^T) @ V ----------
__global__ __launch_bounds__(32)
void k_attn_out(const __bf16* __restrict__ Qp, const __bf16* __restrict__ Kp,
                const __bf16* __restrict__ VpT, const float* __restrict__ stats,
                float* __restrict__ O, int Nq, int Nk)
{
  __shared__ __align__(16) __bf16 P[16 * 40];
  const int lane = threadIdx.x;
  const int col = lane & 15;
  const int halfId = lane >> 4;
  const long r = blockIdx.y;
  const int qt = blockIdx.x;

  bf16x16 qa[3];
  const __bf16* qrow = Qp + ((r * Nq) + qt * 16 + col) * 96;
#pragma unroll
  for (int f = 0; f < 3; ++f) qa[f] = load_fragA(qrow, halfId, f * 32);

  float mx[8], il[8];
#pragma unroll
  for (int j = 0; j < 8; ++j) {
    long q = (long)qt * 16 + halfId * 8 + j;
    mx[j] = stats[(r * Nq + q) * 2 + 0];
    il[j] = 1.f / stats[(r * Nq + q) * 2 + 1];
  }

  f32x8 acc[6];
#pragma unroll
  for (int d6 = 0; d6 < 6; ++d6) acc[d6] = (f32x8){};

  for (int kt = 0; kt < Nk; kt += 32) {
    const __bf16* k0row = Kp + ((r * Nk) + kt + col) * 96;
    const __bf16* k1row = k0row + 16 * 96;
    f32x8 s0 = {}, s1 = {};
#pragma unroll
    for (int f = 0; f < 3; ++f) {
      s0 = wmma_bf16(qa[f], load_fragB(k0row, halfId, f * 32), s0);
      s1 = wmma_bf16(qa[f], load_fragB(k1row, halfId, f * 32), s1);
    }
#pragma unroll
    for (int j = 0; j < 8; ++j) {
      int m = halfId * 8 + j;
      P[m * 40 + col]      = (__bf16)(__expf(s0[j] - mx[j]) * il[j]);
      P[m * 40 + 16 + col] = (__bf16)(__expf(s1[j] - mx[j]) * il[j]);
    }
    __syncthreads();
    bf16x16 pf = load_fragA(P + col * 40, halfId, 0);
#pragma unroll
    for (int d6 = 0; d6 < 6; ++d6) {
      const __bf16* vrow = VpT + ((r * 96) + d6 * 16 + col) * Nk + kt;
      acc[d6] = wmma_bf16(pf, load_fragB(vrow, halfId, 0), acc[d6]);
    }
    __syncthreads();
  }
#pragma unroll
  for (int d6 = 0; d6 < 6; ++d6)
#pragma unroll
    for (int j = 0; j < 8; ++j) {
      long m = (long)qt * 16 + halfId * 8 + j;
      O[(r * Nq + m) * 96 + d6 * 16 + col] = acc[d6][j];
    }
}

// ---------- unpack O [R,Nq,96] -> [B*Nq, 1152] ----------
__global__ void k_unpack(const float* __restrict__ O, float* __restrict__ dst,
                         int H, int Nq, long total)
{
  long i = (long)blockIdx.x * blockDim.x + threadIdx.x;
  if (i >= total) return;
  int d = (int)(i % 1152);
  long rest = i / 1152;
  int q = (int)(rest % Nq);
  long b = rest / Nq;
  int h = d / 72, dd = d - h * 72;
  dst[i] = O[(((b * H) + h) * (long)Nq + q) * 96 + dd];
}

// ---------- RMSNorm ----------
__global__ __launch_bounds__(256)
void k_rmsnorm(const float* __restrict__ src, const float* __restrict__ w,
               float* __restrict__ dst, int D)
{
  __shared__ float sm[256];
  __shared__ float sinv;
  const long row = blockIdx.x;
  float ss = 0.f;
  for (int c = threadIdx.x; c < D; c += 256) { float v = src[row * D + c]; ss += v * v; }
  sm[threadIdx.x] = ss; __syncthreads();
  for (int s = 128; s > 0; s >>= 1) {
    if ((int)threadIdx.x < s) sm[threadIdx.x] += sm[threadIdx.x + s];
    __syncthreads();
  }
  if (threadIdx.x == 0) sinv = rsqrtf(sm[0] / (float)D + 1.1920929e-07f);
  __syncthreads();
  float inv = sinv;
  for (int c = threadIdx.x; c < D; c += 256) dst[row * D + c] = src[row * D + c] * inv * w[c];
}

// ---------- stable descending bitonic argsort per batch ----------
__global__ __launch_bounds__(1024)
void k_sort(const float* __restrict__ s2g, int* __restrict__ order, int N)
{
  __shared__ float key[1024];
  __shared__ int idx[1024];
  const int t = threadIdx.x;
  const long b = blockIdx.x;
  key[t] = s2g[b * N + t]; idx[t] = t;
  __syncthreads();
  for (int k = 2; k <= N; k <<= 1) {
    for (int j = k >> 1; j > 0; j >>= 1) {
      int ixj = t ^ j;
      if (ixj > t) {
        float k1 = key[t], k2 = key[ixj];
        int i1 = idx[t], i2 = idx[ixj];
        bool before = (k1 > k2) || (k1 == k2 && i1 < i2);
        bool up = ((t & k) == 0);
        if (up ? !before : before) { key[t] = k2; key[ixj] = k1; idx[t] = i2; idx[ixj] = i1; }
      }
      __syncthreads();
    }
  }
  order[b * N + t] = idx[t];
}

// ---------- gather rows (bf16 out: feeds bf16 GEMMs) ----------
__global__ void k_gather(const float* __restrict__ x, const int* __restrict__ order,
                         __bf16* __restrict__ hidden, __bf16* __restrict__ qset,
                         int N, int D, int ksel, int rowsH)
{
  const long b = blockIdx.y;
  const int pos = blockIdx.x;
  const int src = order[b * N + pos];
  const float* s = x + (b * N + src) * (long)D;
  __bf16* d = (pos < ksel) ? hidden + (b * rowsH + pos) * (long)D
                           : qset + (b * (long)(N - ksel) + (pos - ksel)) * (long)D;
  for (int c = threadIdx.x; c < D; c += blockDim.x) d[c] = (__bf16)s[c];
}

__global__ void k_copy_rows(const float* __restrict__ src, __bf16* __restrict__ hidden,
                            int rowsPerB, int D, int rowsH, int off)
{
  const long b = blockIdx.y;
  const int rr = blockIdx.x;
  const float* s = src + (b * rowsPerB + rr) * (long)D;
  __bf16* d = hidden + (b * rowsH + off + rr) * (long)D;
  for (int c = threadIdx.x; c < D; c += blockDim.x) d[c] = (__bf16)s[c];
}

__global__ void k_zero(float* __restrict__ p, long n)
{
  long i = (long)blockIdx.x * blockDim.x + threadIdx.x;
  if (i < n) p[i] = 0.f;
}

// ============================= host side =============================
extern "C" void kernel_launch(void* const* d_in, const int* in_sizes, int n_in,
                              void* d_out, int out_size, void* d_ws, size_t ws_size,
                              hipStream_t stream)
{
  (void)in_sizes; (void)n_in; (void)out_size; (void)ws_size;
  const int B = 8, N = 1024, D = 1152, H = 16, L = 64, Hff = 4608, Dt = 4096;
  const int KSEL = 256, NQSET = N - KSEL, NH = KSEL + L;
  const float qscale = 1.0f / sqrtf(72.0f);

  const float* x         = (const float*)d_in[0];
  const float* sel_in_w  = (const float*)d_in[1];
  const float* sel_in_b  = (const float*)d_in[2];
  const float* lqv       = (const float*)d_in[3];
  const float* mhca_in_w = (const float*)d_in[4];
  const float* mhca_in_b = (const float*)d_in[5];
  const float* mhca_out_w= (const float*)d_in[6];
  const float* mhca_out_b= (const float*)d_in[7];
  const float* mha_in_w  = (const float*)d_in[8];
  const float* mha_in_b  = (const float*)d_in[9];
  const float* mha_out_w = (const float*)d_in[10];
  const float* mha_out_b = (const float*)d_in[11];
  const float* norm1_w   = (const float*)d_in[12];
  const float* norm2_w   = (const float*)d_in[13];
  const float* mlp_w1    = (const float*)d_in[14];
  const float* mlp_b1    = (const float*)d_in[15];
  const float* mlp_w2    = (const float*)d_in[16];
  const float* mlp_b2    = (const float*)d_in[17];
  const float* proj_w    = (const float*)d_in[18];

  char* ws = (char*)d_ws;
  const size_t MB = 1ull << 20;
  // region A (reused across phases)
  float*  qk_sel = (float*)(ws);           // [8192,2304] f32
  float*  kv_c   = (float*)(ws);           // [6144,2304] f32 (after qk dead)
  float*  qkv_m  = (float*)(ws);           // [512,3456]  f32
  float*  mlp1   = (float*)(ws + 8 * MB);  // [512,4608]  f32
  // region B (bf16 attention packs, reused)
  __bf16* Qp_sel = (__bf16*)(ws + 80 * MB);
  __bf16* Kp_sel = (__bf16*)(ws + 106 * MB);
  __bf16* Kp_c   = (__bf16*)(ws + 80 * MB);
  __bf16* VpT_c  = (__bf16*)(ws + 100 * MB);
  __bf16* Qp_m   = (__bf16*)(ws + 120 * MB);
  __bf16* Kp_m   = (__bf16*)(ws + 122 * MB);
  __bf16* VpT_m  = (__bf16*)(ws + 124 * MB);
  // region C
  float*  stats_sel = (float*)(ws + 136 * MB);
  float*  s2        = (float*)(ws + 138 * MB);
  int*    order     = (int*)  (ws + 139 * MB);
  __bf16* qset_bf   = (__bf16*)(ws + 140 * MB);   // [6144,1152] bf16
  __bf16* hidden_bf = (__bf16*)(ws + 169 * MB);   // [2560,1152] bf16
  float*  q_lqv     = (float*)(ws + 181 * MB);
  __bf16* Qp_c      = (__bf16*)(ws + 182 * MB);
  float*  stats_c   = (float*)(ws + 183 * MB);
  float*  O_c       = (float*)(ws + 184 * MB);
  float*  attn_c    = (float*)(ws + 188 * MB);
  float*  xc        = (float*)(ws + 191 * MB);
  float*  xn        = (float*)(ws + 194 * MB);
  float*  stats_m   = (float*)(ws + 197 * MB);
  float*  O_m       = (float*)(ws + 198 * MB);
  float*  attn_m    = (float*)(ws + 202 * MB);
  float*  xc2       = (float*)(ws + 205 * MB);
  float*  xn2       = (float*)(ws + 208 * MB);
  float*  cond      = (float*)(ws + 211 * MB);
  // region D: bf16 operand copies for async-LDS GEMMs
  __bf16* x_bf      = (__bf16*)(ws + 214 * MB);   // [8192,1152]
  __bf16* sel_w_bf  = (__bf16*)(ws + 233 * MB);   // [2304,1152]
  __bf16* proj_w_bf = (__bf16*)(ws + 239 * MB);   // [4096,1152]
  __bf16* kv_w_bf   = (__bf16*)(ws + 249 * MB);   // [2304,1152]

  dim3 b256(256), b32(32), b128(128), b1024(1024);
  auto blocks = [](long n) { return dim3((unsigned)((n + 255) / 256)); };

  // 0. one-time bf16 operand conversion (halves L2 traffic for re-read tiles)
  long nx = (long)B * N * D, nsw = (long)2 * D * D, npw = (long)Dt * D;
  k_cvt<<<blocks(nx),  b256, 0, stream>>>(x, x_bf, nx);
  k_cvt<<<blocks(nsw), b256, 0, stream>>>(sel_in_w, sel_w_bf, nsw);
  k_cvt<<<blocks(nsw), b256, 0, stream>>>(mhca_in_w + (size_t)D * D, kv_w_bf, nsw);
  k_cvt<<<blocks(npw), b256, 0, stream>>>(proj_w, proj_w_bf, npw);

  // 1. selector q,k projection (async-LDS bf16 GEMM): [8192,2304]
  k_gemm_bf16<<<dim3((2 * D) / 64, (B * N) / 128), b256, 0, stream>>>(
      x_bf, sel_w_bf, sel_in_b, qk_sel, 2 * D, D);
  // 2-3. pack q (scaled) and k
  long tQ = (long)B * H * N * 96;
  k_pack<<<blocks(tQ), b256, 0, stream>>>(qk_sel, Qp_sel, H, N, 2 * D, (long)N * 2 * D, 0, qscale, tQ);
  k_pack<<<blocks(tQ), b256, 0, stream>>>(qk_sel, Kp_sel, H, N, 2 * D, (long)N * 2 * D, D, 1.f, tQ);
  // 4. flash stats
  k_attn_stats<<<dim3(N / 16, B * H), b32, 0, stream>>>(Qp_sel, Kp_sel, stats_sel, N, N);
  // 5-6. s2 accumulation
  k_zero<<<blocks(B * N), b256, 0, stream>>>(s2, (long)B * N);
  k_sel_s2<<<dim3(N / 16, B * H), b32, 0, stream>>>(Qp_sel, Kp_sel, stats_sel, s2, N, N, H);
  // 7. stable descending argsort per batch
  k_sort<<<dim3(B), b1024, 0, stream>>>(s2, order, N);
  // 8. gather k_set (-> hidden rows 0..255) and q_set (both bf16)
  k_gather<<<dim3(N, B), b128, 0, stream>>>(x, order, hidden_bf, qset_bf, N, D, KSEL, NH);
  // 9. MHCA latent query projection (64 rows, f32 path)
  k_gemm<<<dim3(D / 64, 1), b256, 0, stream>>>(
      lqv, mhca_in_w, mhca_in_b, nullptr, q_lqv, L, D, D, 0);
  // 10. MHCA k,v projection over q_set (async-LDS bf16 GEMM)
  k_gemm_bf16<<<dim3((2 * D) / 64, (B * NQSET) / 128), b256, 0, stream>>>(
      qset_bf, kv_w_bf, mhca_in_b + D, kv_c, 2 * D, D);
  // 11-13. packs
  long tQc = (long)B * H * L * 96;
  k_pack<<<blocks(tQc), b256, 0, stream>>>(q_lqv, Qp_c, H, L, D, 0L, 0, qscale, tQc);
  long tKc = (long)B * H * NQSET * 96;
  k_pack<<<blocks(tKc), b256, 0, stream>>>(kv_c, Kp_c, H, NQSET, 2 * D, (long)NQSET * 2 * D, 0, 1.f, tKc);
  k_pack_vT<<<blocks(tKc), b256, 0, stream>>>(kv_c, VpT_c, H, NQSET, 2 * D, (long)NQSET * 2 * D, D, tKc);
  // 14-15. MHCA attention
  k_attn_stats<<<dim3(L / 16, B * H), b32, 0, stream>>>(Qp_c, Kp_c, stats_c, L, NQSET);
  k_attn_out<<<dim3(L / 16, B * H), b32, 0, stream>>>(Qp_c, Kp_c, VpT_c, stats_c, O_c, L, NQSET);
  // 16-17. unpack + out projection
  long tU = (long)B * L * D;
  k_unpack<<<blocks(tU), b256, 0, stream>>>(O_c, attn_c, H, L, tU);
  k_gemm<<<dim3(D / 64, (B * L) / 64), b256, 0, stream>>>(
      attn_c, mhca_out_w, mhca_out_b, nullptr, xc, B * L, D, D, 0);
  // 18. rmsnorm 1
  k_rmsnorm<<<dim3(B * L), b256, 0, stream>>>(xc, norm1_w, xn, D);
  // 19. self-MHA qkv projection
  k_gemm<<<dim3((3 * D) / 64, (B * L) / 64), b256, 0, stream>>>(
      xn, mha_in_w, mha_in_b, nullptr, qkv_m, B * L, 3 * D, D, 0);
  // 20. packs
  long tM = (long)B * H * L * 96;
  k_pack<<<blocks(tM), b256, 0, stream>>>(qkv_m, Qp_m, H, L, 3 * D, (long)L * 3 * D, 0, qscale, tM);
  k_pack<<<blocks(tM), b256, 0, stream>>>(qkv_m, Kp_m, H, L, 3 * D, (long)L * 3 * D, D, 1.f, tM);
  k_pack_vT<<<blocks(tM), b256, 0, stream>>>(qkv_m, VpT_m, H, L, 3 * D, (long)L * 3 * D, 2 * D, tM);
  // 21-22. self-MHA attention
  k_attn_stats<<<dim3(L / 16, B * H), b32, 0, stream>>>(Qp_m, Kp_m, stats_m, L, L);
  k_attn_out<<<dim3(L / 16, B * H), b32, 0, stream>>>(Qp_m, Kp_m, VpT_m, stats_m, O_m, L, L);
  // 23-24. unpack + out projection with residual (xc2 = xn + attn_out)
  k_unpack<<<blocks(tU), b256, 0, stream>>>(O_m, attn_m, H, L, tU);
  k_gemm<<<dim3(D / 64, (B * L) / 64), b256, 0, stream>>>(
      attn_m, mha_out_w, mha_out_b, xn, xc2, B * L, D, D, 0);
  // 25. rmsnorm 2
  k_rmsnorm<<<dim3(B * L), b256, 0, stream>>>(xc2, norm2_w, xn2, D);
  // 26-27. MLP with GELU, residual (cond = xn2 + ff)
  k_gemm<<<dim3(Hff / 64, (B * L) / 64), b256, 0, stream>>>(
      xn2, mlp_w1, mlp_b1, nullptr, mlp1, B * L, Hff, D, 1);
  k_gemm<<<dim3(D / 64, (B * L) / 64), b256, 0, stream>>>(
      mlp1, mlp_w2, mlp_b2, xn2, cond, B * L, D, Hff, 0);
  // 28. place condensed rows into hidden rows 256..319 (bf16)
  k_copy_rows<<<dim3(L, B), b128, 0, stream>>>(cond, hidden_bf, L, D, NH, KSEL);
  // 29. final projection (async-LDS bf16 GEMM) -> d_out [2560, 4096]
  k_gemm_bf16<<<dim3(Dt / 64, (B * NH) / 128), b256, 0, stream>>>(
      hidden_bf, proj_w_bf, nullptr, (float*)d_out, Dt, D);
}
